// LaneAwareProbabilityLearning_60327110639831
// MI455X (gfx1250) — matrix operand order repaired and checked by reading, not compiled, not run on hardware
//
#include <hip/hip_runtime.h>
#include <hip/hip_bf16.h>

// ---------------------------------------------------------------------------
// LaneAwareProbabilityLearning for MI455X (gfx1250, wave32, WMMA).
// All dense GEMMs -> v_wmma_f32_16x16x32_f16 (f16 in, f32 accumulate).
// Norms / scan / softmax reductions stay in f32.
// ---------------------------------------------------------------------------

typedef __attribute__((ext_vector_type(16))) _Float16 v16h;
typedef __attribute__((ext_vector_type(8)))  _Float16 v8h;
typedef __attribute__((ext_vector_type(8)))  float    v8f;

#define DMODEL 512
#define DINNER 1024
#define DSTATE 16
#define DTRANK 32
#define NBLK   4
#define KCONV  4
#define NHEAD  4
#define DHEAD  128
#define NCLS   6
#define BB     8
#define LQ     512   // lane sequence length
#define LK     256   // agent sequence length

// ------------------------------- converts ----------------------------------
__global__ void cvt_f16_k(const float* __restrict__ in, _Float16* __restrict__ out, int n) {
    int t = blockIdx.x * blockDim.x + threadIdx.x;
    if (t < n) out[t] = (_Float16)in[t];
}

// in: f32 [K,N] row-major -> out: f16 [N,K] row-major (B-transposed for WMMA)
__global__ void cvt_transpose_f16_k(const float* __restrict__ in, _Float16* __restrict__ out,
                                    int K, int N) {
    int t = blockIdx.x * blockDim.x + threadIdx.x;
    if (t >= K * N) return;
    int k = t % K, n = t / K;
    out[t] = (_Float16)in[(long)k * N + n];
}

// ------------------------------- WMMA GEMM ---------------------------------
// C[M,N] = act( scale * (A[M,K] x B[K,N]) + bias ) (+ C_old if accum)
// A: f16 row-major, lda; BT: f16 [N,K] row-major, ldbt. Batched via blockIdx.z.
// One wave per 16x16 C tile, 4 waves (4 N-tiles) per block. K % 32 == 0.
__global__ __launch_bounds__(128)
void wmma_gemm_k(const _Float16* __restrict__ A, const _Float16* __restrict__ BT,
                 const float* __restrict__ bias, float* __restrict__ C,
                 _Float16* __restrict__ C16,
                 int M, int N, int K, int lda, int ldbt, int ldc,
                 int sA, int sB, int sC, float scale, int accum, int act) {
    (void)M;
    const int wave = threadIdx.x >> 5;
    const int lane = threadIdx.x & 31;
    const int n0 = (blockIdx.x * 4 + wave) * 16;
    const int m0 = blockIdx.y * 16;
    if (n0 >= N) return;                 // wave-uniform: EXEC stays all-ones for WMMA

    const int z = blockIdx.z;
    A  += (long)z * sA;
    BT += (long)z * sB;

    const int g    = lane >> 4;          // 0: low half, 1: high half of K-subtile
    const int mrow = m0 + (lane & 15);   // A row for this lane
    const int ncol = n0 + (lane & 15);   // B column / C column for this lane

    v8f acc = {};
    for (int k = 0; k < K; k += 32) {
        // A frag (lanes 0-15: K=k..k+15, lanes 16-31: K=k+16..k+31, row = mrow)
        const _Float16* ap = A + (long)mrow * lda + k + g * 16;
        v16h a = *(const v16h*)(ap);
        // B frag (lanes 0-15: K=k..k+15, lanes 16-31: K=k+16..k+31, col = ncol)
        v16h b = *(const v16h*)(BT + (long)ncol * ldbt + k + g * 16);
        acc = __builtin_amdgcn_wmma_f32_16x16x32_f16(false, a, false, b,
                                                     (short)0, acc, false, false);
    }

    float* Cb = C ? C + (long)z * sC : nullptr;
    _Float16* C16b = C16 ? C16 + (long)z * sC : nullptr;
#pragma unroll
    for (int r = 0; r < 8; ++r) {
        int m = m0 + g * 8 + r;          // ISA C/D layout: VGPR r -> row r (+8 for hi lanes)
        float v = acc[r] * scale;
        if (bias) v += bias[ncol];
        if (act == 1)       v = v / (1.f + __expf(-v));                    // silu
        else if (act == 2)  v = (v > 20.f) ? v : __logf(1.f + __expf(v));  // softplus
        long idx = (long)m * ldc + ncol;
        if (accum && Cb) v += Cb[idx];
        if (Cb)   Cb[idx]   = v;
        if (C16b) C16b[idx] = (_Float16)v;
    }
}

// ------------------------------- RMSNorm -----------------------------------
__global__ void rmsnorm_f16_k(const float* __restrict__ x, const float* __restrict__ w,
                              _Float16* __restrict__ out, int rows, int D) {
    int wave = threadIdx.x >> 5, lane = threadIdx.x & 31;
    int row = blockIdx.x * (blockDim.x >> 5) + wave;
    if (row >= rows) return;
    const float* xr = x + (long)row * D;
    float ss = 0.f;
    for (int i = lane; i < D; i += 32) { float v = xr[i]; ss += v * v; }
    for (int o = 16; o > 0; o >>= 1) ss += __shfl_xor(ss, o, 32);
    float sc = rsqrtf(ss / (float)D + 1e-5f);
    _Float16* orow = out + (long)row * D;
    for (int i = lane; i < D; i += 32) orow[i] = (_Float16)(xr[i] * sc * w[i]);
}

// --------------------- causal depthwise conv (K=4) + SiLU ------------------
// xz: f16 (B, L, 2*Din); conv over the first Din channels along L.
__global__ void conv_silu_k(const _Float16* __restrict__ xz, const float* __restrict__ w,
                            const float* __restrict__ b, float* __restrict__ xc,
                            _Float16* __restrict__ xc16, int B, int L, int Din) {
    int t = blockIdx.x * blockDim.x + threadIdx.x;
    int total = B * L * Din;
    if (t >= total) return;
    int d = t % Din, bl = t / Din, l = bl % L, bb = bl / L;
    float acc = b[d];
#pragma unroll
    for (int j = 0; j < KCONV; ++j) {
        int ls = l - (KCONV - 1) + j;
        if (ls >= 0) acc += w[d * KCONV + j] * (float)xz[((long)bb * L + ls) * (2 * Din) + d];
    }
    float s = acc / (1.f + __expf(-acc));
    xc[t] = s;
    xc16[t] = (_Float16)s;
}

// -------------------------- selective scan ---------------------------------
// One thread per (batch, channel); sequential over L with 16-wide state.
__global__ void scan_k(const float* __restrict__ u, const float* __restrict__ delta,
                       const _Float16* __restrict__ dbl, const float* __restrict__ A_log,
                       const float* __restrict__ Dskip, float* __restrict__ y,
                       int B, int L, int Din) {
    int t = blockIdx.x * blockDim.x + threadIdx.x;
    if (t >= B * Din) return;
    int b = t / Din, d = t % Din;
    float Ar[DSTATE];
#pragma unroll
    for (int s = 0; s < DSTATE; ++s) Ar[s] = -__expf(A_log[(long)d * DSTATE + s]);
    float h[DSTATE];
#pragma unroll
    for (int s = 0; s < DSTATE; ++s) h[s] = 0.f;
    float Dd = Dskip[d];
    for (int l = 0; l < L; ++l) {
        long row = (long)b * L + l;
        float dl = delta[row * Din + d];
        float uu = u[row * Din + d];
        float du = dl * uu;
        const _Float16* bl = dbl + row * (DTRANK + 2 * DSTATE);
        float acc = 0.f;
#pragma unroll
        for (int s = 0; s < DSTATE; ++s) {
            float dA = __expf(dl * Ar[s]);
            h[s] = dA * h[s] + du * (float)bl[DTRANK + s];
            acc += h[s] * (float)bl[DTRANK + DSTATE + s];
        }
        y[row * Din + d] = acc + uu * Dd;
    }
}

// --------------------------- y * silu(z) gate ------------------------------
__global__ void gate_k(const float* __restrict__ y, const _Float16* __restrict__ xz,
                       _Float16* __restrict__ g, int total, int Din) {
    int t = blockIdx.x * blockDim.x + threadIdx.x;
    if (t >= total) return;
    int d = t % Din; long bl = t / Din;
    float z = (float)xz[bl * (2 * Din) + Din + d];
    float s = z / (1.f + __expf(-z));
    g[t] = (_Float16)(y[t] * s);
}

// ------------------------------- softmax -----------------------------------
__global__ void softmax_rows_k(const _Float16* __restrict__ s, _Float16* __restrict__ p,
                               int rows, int cols) {
    int wave = threadIdx.x >> 5, lane = threadIdx.x & 31;
    int row = blockIdx.x * (blockDim.x >> 5) + wave;
    if (row >= rows) return;
    const _Float16* sr = s + (long)row * cols;
    float m = -1e30f;
    for (int i = lane; i < cols; i += 32) m = fmaxf(m, (float)sr[i]);
    for (int o = 16; o > 0; o >>= 1) m = fmaxf(m, __shfl_xor(m, o, 32));
    float sum = 0.f;
    for (int i = lane; i < cols; i += 32) sum += __expf((float)sr[i] - m);
    for (int o = 16; o > 0; o >>= 1) sum += __shfl_xor(sum, o, 32);
    float inv = 1.f / sum;
    _Float16* pr = p + (long)row * cols;
    for (int i = lane; i < cols; i += 32) pr[i] = (_Float16)(__expf((float)sr[i] - m) * inv);
}

// ------------------------------ head repacks -------------------------------
// (B, L, H*dh) -> (B*H, L, dh)
__global__ void repack_heads_k(const _Float16* __restrict__ in, _Float16* __restrict__ out,
                               int B, int L, int H, int dh) {
    int t = blockIdx.x * blockDim.x + threadIdx.x;
    if (t >= B * L * H * dh) return;
    int d = t % dh; int r = t / dh; int l = r % L; int bh = r / L;
    int h = bh % H; int b = bh / H;
    out[t] = in[((long)b * L + l) * (H * dh) + h * dh + d];
}
// (B, L, H*dh) -> (B*H, dh, L)   (V transposed for P@V)
__global__ void repack_heads_T_k(const _Float16* __restrict__ in, _Float16* __restrict__ out,
                                 int B, int L, int H, int dh) {
    int t = blockIdx.x * blockDim.x + threadIdx.x;
    if (t >= B * L * H * dh) return;
    int l = t % L; int r = t / L; int d = r % dh; int bh = r / dh;
    int h = bh % H; int b = bh / H;
    out[t] = in[((long)b * L + l) * (H * dh) + h * dh + d];
}
// (B*H, L, dh) -> (B, L, H*dh)
__global__ void unpack_heads_k(const _Float16* __restrict__ in, _Float16* __restrict__ out,
                               int B, int L, int H, int dh) {
    int t = blockIdx.x * blockDim.x + threadIdx.x;
    if (t >= B * L * H * dh) return;
    int c = t % (H * dh); int l = (t / (H * dh)) % L; int b = t / (H * dh * L);
    int h = c / dh; int d = c % dh;
    out[t] = in[(((long)b * H + h) * L + l) * dh + d];
}

// --------------------- final 512x6 head + argmax ---------------------------
__global__ void head_k(const _Float16* __restrict__ mid, const float* __restrict__ W,
                       const float* __restrict__ bias, float* __restrict__ probs,
                       float* __restrict__ preds, int rows) {
    int r = blockIdx.x * blockDim.x + threadIdx.x;
    if (r >= rows) return;
    const _Float16* mr = mid + (long)r * DMODEL;
    float best = -1e30f; int bi = 0;
    for (int n = 0; n < NCLS; ++n) {
        float s = bias[n];
        for (int k = 0; k < DMODEL; ++k) s += (float)mr[k] * W[k * NCLS + n];
        probs[(long)r * NCLS + n] = s;
        if (s > best) { best = s; bi = n; }
    }
    preds[r] = (float)bi;
}

// ===========================================================================
extern "C" void kernel_launch(void* const* d_in, const int* in_sizes, int n_in,
                              void* d_out, int out_size, void* d_ws, size_t ws_size,
                              hipStream_t stream) {
    (void)in_sizes; (void)n_in; (void)out_size; (void)ws_size;

    const float* agent      = (const float*)d_in[0];
    const float* lanes      = (const float*)d_in[1];
    const float* lane_in_W  = (const float*)d_in[2];
    const float* lane_in_b  = (const float*)d_in[3];
    const float* norm_w     = (const float*)d_in[4];
    const float* in_proj_W  = (const float*)d_in[5];
    const float* conv_w     = (const float*)d_in[6];
    const float* conv_b     = (const float*)d_in[7];
    const float* x_proj_W   = (const float*)d_in[8];
    const float* dt_proj_W  = (const float*)d_in[9];
    const float* dt_proj_b  = (const float*)d_in[10];
    const float* A_log      = (const float*)d_in[11];
    const float* D_skip     = (const float*)d_in[12];
    const float* out_proj_W = (const float*)d_in[13];
    const float* final_nw   = (const float*)d_in[14];
    const float* Wq = (const float*)d_in[15];
    const float* Wk = (const float*)d_in[16];
    const float* Wv = (const float*)d_in[17];
    const float* Wo = (const float*)d_in[18];
    const float* bq = (const float*)d_in[19];
    const float* bk = (const float*)d_in[20];
    const float* bv = (const float*)d_in[21];
    const float* bo = (const float*)d_in[22];
    const float* lin_in_W  = (const float*)d_in[23];
    const float* lin_in_b  = (const float*)d_in[24];
    const float* lin_out_W = (const float*)d_in[25];
    const float* lin_out_b = (const float*)d_in[26];

    const int Mx = BB * LQ;   // 4096 lane rows
    const int Ma = BB * LK;   // 2048 agent rows
    const int DBL = DTRANK + 2 * DSTATE;  // 64

    // ---------------- workspace arena ----------------
    char* ws = (char*)d_ws;
    size_t off = 0;
    auto alloc = [&](size_t bytes) -> void* {
        size_t cur = (off + 255) & ~(size_t)255;
        off = cur + bytes;
        return (void*)(ws + cur);
    };
    // persistent region (f16 transposed weights + buffers that must survive aliasing)
    _Float16* laneinT  = (_Float16*)alloc((size_t)DMODEL * 128 * 2);
    _Float16* inprojT  = (_Float16*)alloc((size_t)NBLK * 2 * DINNER * DMODEL * 2);
    _Float16* xprojT   = (_Float16*)alloc((size_t)NBLK * DBL * DINNER * 2);
    _Float16* dtprojT  = (_Float16*)alloc((size_t)NBLK * DINNER * DTRANK * 2);
    _Float16* outprojT = (_Float16*)alloc((size_t)NBLK * DMODEL * DINNER * 2);
    _Float16* WqT   = (_Float16*)alloc((size_t)DMODEL * DMODEL * 2);
    _Float16* WkT   = (_Float16*)alloc((size_t)DMODEL * DMODEL * 2);
    _Float16* WvT   = (_Float16*)alloc((size_t)DMODEL * DMODEL * 2);
    _Float16* WoT   = (_Float16*)alloc((size_t)DMODEL * DMODEL * 2);
    _Float16* linT  = (_Float16*)alloc((size_t)DMODEL * DMODEL * 2);
    _Float16* enh16 = (_Float16*)alloc((size_t)Mx * DMODEL * 2);
    _Float16* ag16  = (_Float16*)alloc((size_t)Ma * DMODEL * 2);
    size_t mark = off;
    // mamba region
    float*    x     = (float*)   alloc((size_t)Mx * DMODEL * 4);
    _Float16* lf16  = (_Float16*)alloc((size_t)Mx * 128 * 2);
    _Float16* h16   = (_Float16*)alloc((size_t)Mx * DMODEL * 2);
    _Float16* xz16  = (_Float16*)alloc((size_t)Mx * 2 * DINNER * 2);
    float*    xc    = (float*)   alloc((size_t)Mx * DINNER * 4);
    _Float16* xc16  = (_Float16*)alloc((size_t)Mx * DINNER * 2);
    _Float16* dbl16 = (_Float16*)alloc((size_t)Mx * DBL * 2);
    float*    delta = (float*)   alloc((size_t)Mx * DINNER * 4);
    float*    yb    = (float*)   alloc((size_t)Mx * DINNER * 4);
    _Float16* g16   = (_Float16*)alloc((size_t)Mx * DINNER * 2);
    // attention region (aliases the mamba region; mamba is dead once enh16 exists)
    off = mark;
    _Float16* qf16  = (_Float16*)alloc((size_t)Mx * DMODEL * 2);
    _Float16* kf16  = (_Float16*)alloc((size_t)Ma * DMODEL * 2);
    _Float16* vf16  = (_Float16*)alloc((size_t)Ma * DMODEL * 2);
    _Float16* qh    = (_Float16*)alloc((size_t)BB * NHEAD * LQ * DHEAD * 2);
    _Float16* kh    = (_Float16*)alloc((size_t)BB * NHEAD * LK * DHEAD * 2);
    _Float16* vT    = (_Float16*)alloc((size_t)BB * NHEAD * DHEAD * LK * 2);
    _Float16* s16   = (_Float16*)alloc((size_t)BB * NHEAD * LQ * LK * 2);
    _Float16* p16   = (_Float16*)alloc((size_t)BB * NHEAD * LQ * LK * 2);
    _Float16* oh16  = (_Float16*)alloc((size_t)BB * NHEAD * LQ * DHEAD * 2);
    _Float16* o16   = (_Float16*)alloc((size_t)Mx * DMODEL * 2);
    _Float16* ao16  = (_Float16*)alloc((size_t)Mx * DMODEL * 2);
    _Float16* mid16 = (_Float16*)alloc((size_t)Mx * DMODEL * 2);

    auto cvt = [&](const float* in, _Float16* out, size_t n) {
        cvt_f16_k<<<(unsigned)((n + 255) / 256), 256, 0, stream>>>(in, out, (int)n);
    };
    auto cvtT = [&](const float* in, _Float16* out, int K, int N) {
        cvt_transpose_f16_k<<<(unsigned)(((size_t)K * N + 255) / 256), 256, 0, stream>>>(in, out, K, N);
    };
    auto gemm = [&](const _Float16* A, const _Float16* BT, const float* bias,
                    float* C, _Float16* C16, int M, int N, int K,
                    int lda, int ldbt, int ldc, int batches,
                    int sA, int sB, int sC, float scale, int accum, int act) {
        dim3 grid((N + 63) / 64, M / 16, batches), blk(128);
        wmma_gemm_k<<<grid, blk, 0, stream>>>(A, BT, bias, C, C16, M, N, K,
                                              lda, ldbt, ldc, sA, sB, sC, scale, accum, act);
    };

    // ---------------- weight + input conversion ----------------
    cvt(lanes, lf16, (size_t)Mx * 128);
    cvt(agent, ag16, (size_t)Ma * DMODEL);
    cvtT(lane_in_W, laneinT, 128, DMODEL);
    for (int i = 0; i < NBLK; ++i) {
        cvtT(in_proj_W  + (size_t)i * DMODEL * 2 * DINNER, inprojT  + (size_t)i * 2 * DINNER * DMODEL, DMODEL, 2 * DINNER);
        cvtT(x_proj_W   + (size_t)i * DINNER * DBL,        xprojT   + (size_t)i * DBL * DINNER,        DINNER, DBL);
        cvtT(dt_proj_W  + (size_t)i * DTRANK * DINNER,     dtprojT  + (size_t)i * DINNER * DTRANK,     DTRANK, DINNER);
        cvtT(out_proj_W + (size_t)i * DINNER * DMODEL,     outprojT + (size_t)i * DMODEL * DINNER,     DINNER, DMODEL);
    }
    cvtT(Wq, WqT, DMODEL, DMODEL);  cvtT(Wk, WkT, DMODEL, DMODEL);
    cvtT(Wv, WvT, DMODEL, DMODEL);  cvtT(Wo, WoT, DMODEL, DMODEL);
    cvtT(lin_in_W, linT, DMODEL, DMODEL);

    // ---------------- Mamba stack ----------------
    // x = lanes @ lane_in_W + b
    gemm(lf16, laneinT, lane_in_b, x, nullptr, Mx, DMODEL, 128, 128, 128, DMODEL,
         1, 0, 0, 0, 1.f, 0, 0);

    const int totI = Mx * DINNER;
    for (int i = 0; i < NBLK; ++i) {
        rmsnorm_f16_k<<<Mx / 8, 256, 0, stream>>>(x, norm_w + (size_t)i * DMODEL, h16, Mx, DMODEL);
        gemm(h16, inprojT + (size_t)i * 2 * DINNER * DMODEL, nullptr, nullptr, xz16,
             Mx, 2 * DINNER, DMODEL, DMODEL, DMODEL, 2 * DINNER, 1, 0, 0, 0, 1.f, 0, 0);
        conv_silu_k<<<totI / 256, 256, 0, stream>>>(xz16, conv_w + (size_t)i * DINNER * KCONV,
                                                    conv_b + (size_t)i * DINNER, xc, xc16, BB, LQ, DINNER);
        gemm(xc16, xprojT + (size_t)i * DBL * DINNER, nullptr, nullptr, dbl16,
             Mx, DBL, DINNER, DINNER, DINNER, DBL, 1, 0, 0, 0, 1.f, 0, 0);
        // delta = softplus(dt @ dt_proj_W + dt_proj_b); dt = first 32 cols of dbl
        gemm(dbl16, dtprojT + (size_t)i * DINNER * DTRANK, dt_proj_b + (size_t)i * DINNER,
             delta, nullptr, Mx, DINNER, DTRANK, DBL, DTRANK, DINNER, 1, 0, 0, 0, 1.f, 0, 2);
        scan_k<<<(BB * DINNER) / 256, 256, 0, stream>>>(xc, delta, dbl16,
                A_log + (size_t)i * DINNER * DSTATE, D_skip + (size_t)i * DINNER, yb, BB, LQ, DINNER);
        gate_k<<<totI / 256, 256, 0, stream>>>(yb, xz16, g16, totI, DINNER);
        // x += (y * silu(z)) @ out_proj_W
        gemm(g16, outprojT + (size_t)i * DMODEL * DINNER, nullptr, x, nullptr,
             Mx, DMODEL, DINNER, DINNER, DINNER, DMODEL, 1, 0, 0, 0, 1.f, 1, 0);
    }
    rmsnorm_f16_k<<<Mx / 8, 256, 0, stream>>>(x, final_nw, enh16, Mx, DMODEL);

    // ---------------- MHA (enhanced as Q, agents as K/V) ----------------
    gemm(enh16, WqT, bq, nullptr, qf16, Mx, DMODEL, DMODEL, DMODEL, DMODEL, DMODEL, 1, 0, 0, 0, 1.f, 0, 0);
    gemm(ag16,  WkT, bk, nullptr, kf16, Ma, DMODEL, DMODEL, DMODEL, DMODEL, DMODEL, 1, 0, 0, 0, 1.f, 0, 0);
    gemm(ag16,  WvT, bv, nullptr, vf16, Ma, DMODEL, DMODEL, DMODEL, DMODEL, DMODEL, 1, 0, 0, 0, 1.f, 0, 0);

    repack_heads_k  <<<(Mx * DMODEL) / 256, 256, 0, stream>>>(qf16, qh, BB, LQ, NHEAD, DHEAD);
    repack_heads_k  <<<(Ma * DMODEL) / 256, 256, 0, stream>>>(kf16, kh, BB, LK, NHEAD, DHEAD);
    repack_heads_T_k<<<(Ma * DMODEL) / 256, 256, 0, stream>>>(vf16, vT, BB, LK, NHEAD, DHEAD);

    // scores = (Q K^T) / sqrt(dh) : batched over B*H = 32
    gemm(qh, kh, nullptr, nullptr, s16, LQ, LK, DHEAD, DHEAD, DHEAD, LK,
         BB * NHEAD, LQ * DHEAD, LK * DHEAD, LQ * LK, 0.0883883476483184f, 0, 0);
    softmax_rows_k<<<(BB * NHEAD * LQ) / 8, 256, 0, stream>>>(s16, p16, BB * NHEAD * LQ, LK);
    // O = P V
    gemm(p16, vT, nullptr, nullptr, oh16, LQ, DHEAD, LK, LK, LK, DHEAD,
         BB * NHEAD, LQ * LK, DHEAD * LK, LQ * DHEAD, 1.f, 0, 0);
    unpack_heads_k<<<(Mx * DMODEL) / 256, 256, 0, stream>>>(oh16, o16, BB, LQ, NHEAD, DHEAD);

    gemm(o16,  WoT,  bo,       nullptr, ao16,  Mx, DMODEL, DMODEL, DMODEL, DMODEL, DMODEL, 1, 0, 0, 0, 1.f, 0, 0);
    gemm(ao16, linT, lin_in_b, nullptr, mid16, Mx, DMODEL, DMODEL, DMODEL, DMODEL, DMODEL, 1, 0, 0, 0, 1.f, 0, 0);

    // ---------------- head: probs (B,L,6) + argmax preds (B,L) ----------------
    float* probs = (float*)d_out;
    float* preds = probs + (size_t)Mx * NCLS;
    head_k<<<Mx / 256, 256, 0, stream>>>(mid16, lin_out_W, lin_out_b, probs, preds, Mx);
}
